// PLoss_my_83133386981798
// MI455X (gfx1250) — matrix operand order repaired and checked by reading, not compiled, not run on hardware
//
#include <hip/hip_runtime.h>
#include <stdint.h>

typedef __attribute__((ext_vector_type(16))) _Float16 v16h;
typedef __attribute__((ext_vector_type(8)))  float    v8f;

#define PL_THREADS 256
#define PL_C 128
#define PL_K 128
#define PL_ROWS_PER_BLOCK 128   // 8 waves * 16 rows

// ---------------------------------------------------------------------------
// Main fused kernel: distances via WMMA + argmin + log-softmax NLL per row.
// Writes one fp32 partial sum per block (deterministic order).
// ---------------------------------------------------------------------------
__global__ __launch_bounds__(PL_THREADS)
void ploss_main_83133386981798(const float* __restrict__ outputs,
                               const int*   __restrict__ labels,
                               const float* __restrict__ protos,
                               const int*   __restrict__ kptr,
                               float*       __restrict__ partial,
                               int nrows)
{
    // LDS: B fragments (f16, WMMA layout) + |p|^2 + argmin keys + wave partials
    __shared__ _Float16           sB[8 * 4 * 32 * 16];   // 32 KB: [jt][kc][lane][h]
    __shared__ float              sP2[PL_K];             // 512 B
    __shared__ unsigned long long sKeys[8 * 16 * 16];    // 16 KB: [wave][row][lane0..15]
    __shared__ float              sPart[8];

    const int tid  = threadIdx.x;
    const int lane = tid & 31;
    const int wave = tid >> 5;

    // ---- Stage protos into exact B-fragment layout (f16) ------------------
    // B fragment for (jt,kc): lane l, half h  <-  proto[16*jt + (l&15)][32*kc + 8*(l>>4) + (h<8 ? h : h+8)]
    for (int t = tid; t < 8 * 4 * 32; t += PL_THREADS) {
        const int jt = t >> 7;
        const int kc = (t >> 5) & 3;
        const int ln = t & 31;
        const int j  = 16 * jt + (ln & 15);
        const int cb = 32 * kc + 8 * (ln >> 4);
        const float* __restrict__ src = protos + (size_t)j * PL_C + cb;
        _Float16* dst = &sB[t * 16];
        #pragma unroll
        for (int h = 0; h < 16; ++h) dst[h] = (_Float16)src[h < 8 ? h : h + 8];
    }
    // |p_j|^2
    if (tid < PL_K) {
        const float* __restrict__ p = protos + (size_t)tid * PL_C;
        float s = 0.f;
        for (int c = 0; c < PL_C; ++c) { const float v = p[c]; s += v * v; }
        sP2[tid] = s;
    }
    __syncthreads();

    // ---- Per-wave 16-row tile ---------------------------------------------
    const int rowBase = blockIdx.x * PL_ROWS_PER_BLOCK + wave * 16;
    const int m  = lane & 15;          // A-matrix row owned by this lane
    const int hi = lane >> 4;          // lane half selects K sub-runs
    int rowA = rowBase + m;
    if (rowA > nrows - 1) rowA = nrows - 1;   // clamp (keeps EXEC full for WMMA)
    const float* __restrict__ rowPtr = outputs + (size_t)rowA * PL_C;

    v8f acc[8];
    #pragma unroll
    for (int jt = 0; jt < 8; ++jt) acc[jt] = (v8f)0.0f;

    float mx = -3.0e38f;   // online logsumexp over this lane's 64 values
    float sm = 0.0f;

    #pragma unroll
    for (int kc = 0; kc < 4; ++kc) {
        // Two contiguous 8-float runs per lane, matching A layout:
        //   h=0..7  -> K = 32*kc + 8*hi + (0..7)
        //   h=8..15 -> K = 32*kc + 16 + 8*hi + (0..7)
        const float4* r1 = (const float4*)(rowPtr + 32 * kc + 8 * hi);
        const float4* r2 = (const float4*)(rowPtr + 32 * kc + 16 + 8 * hi);
        float v[16];
        { float4 a0 = r1[0], a1 = r1[1], b0 = r2[0], b1 = r2[1];
          v[0]=a0.x; v[1]=a0.y; v[2]=a0.z; v[3]=a0.w;
          v[4]=a1.x; v[5]=a1.y; v[6]=a1.z; v[7]=a1.w;
          v[8]=b0.x; v[9]=b0.y; v[10]=b0.z; v[11]=b0.w;
          v[12]=b1.x; v[13]=b1.y; v[14]=b1.z; v[15]=b1.w; }

        // streaming logsumexp update (f32)
        float cm = v[0];
        #pragma unroll
        for (int h = 1; h < 16; ++h) cm = fmaxf(cm, v[h]);
        const float nm = fmaxf(mx, cm);
        sm *= __expf(mx - nm);
        #pragma unroll
        for (int h = 0; h < 16; ++h) sm += __expf(v[h] - nm);
        mx = nm;

        // build f16 A fragment
        v16h a;
        #pragma unroll
        for (int h = 0; h < 16; ++h) a[h] = (_Float16)v[h];

        // 8 proto tiles: acc[jt] += A * B
        #pragma unroll
        for (int jt = 0; jt < 8; ++jt) {
            const v16h b = *(const v16h*)(&sB[((jt * 4 + kc) * 32 + lane) * 16]);
            acc[jt] = __builtin_amdgcn_wmma_f32_16x16x32_f16(
                false, a, false, b, (short)0, acc[jt], false, false);
        }
    }

    // ---- combine partner-lane logsumexp (row m held by lanes m and m+16) --
    {
        const float omx = __shfl_xor(mx, 16);
        const float osm = __shfl_xor(sm, 16);
        const float M = fmaxf(mx, omx);
        sm = __expf(mx - M) * sm + __expf(omx - M) * osm;
        mx = M;
    }
    const float lse = mx + __logf(sm);   // full-row logsumexp (lanes l and l+16 share row l&15)

    // ---- argmin over 128 protos -------------------------------------------
    // D layout: lane l, vgpr r  ->  dot(row r+8*hi, proto jt*16 + (l&15))
    // key = sortable(score)<<32 | idx  => u64-min == argmin w/ first-index ties
    unsigned long long bkey[8];
    #pragma unroll
    for (int r = 0; r < 8; ++r) bkey[r] = ~0ull;
    #pragma unroll
    for (int jt = 0; jt < 8; ++jt) {
        const int   pidx = jt * 16 + (lane & 15);
        const float p2v  = sP2[pidx];
        #pragma unroll
        for (int r = 0; r < 8; ++r) {
            const float score = p2v - 2.0f * acc[jt][r];
            unsigned u = __float_as_uint(score);
            u ^= (unsigned)(((int)u >> 31)) | 0x80000000u;
            const unsigned long long key =
                ((unsigned long long)u << 32) | (unsigned)pidx;
            if (key < bkey[r]) bkey[r] = key;
        }
    }
    #pragma unroll
    for (int r = 0; r < 8; ++r)
        sKeys[(wave * 16 + (r + 8 * hi)) * 16 + (lane & 15)] = bkey[r];
    __syncthreads();

    // ---- per-row finalize: lane < 16 handles row (rowBase + lane) ---------
    float nll = 0.0f;
    if (lane < 16) {
        const int row = rowBase + lane;
        if (row < nrows) {
            unsigned long long best = ~0ull;
            const unsigned long long* kp = &sKeys[(wave * 16 + lane) * 16];
            #pragma unroll
            for (int n = 0; n < 16; ++n) {
                const unsigned long long key = kp[n];
                if (key < best) best = key;
            }
            const int bestIdx = (int)(best & 0xFFFFu);
            const int lab = labels[row];
            const int kk  = kptr[0];
            const int lbl = (lab <= kk - 1) ? lab : bestIdx;
            const float xv = outputs[(size_t)row * PL_C + lbl];
            nll = lse - xv;     // -log_softmax at label
        }
    }

    // ---- deterministic block reduction ------------------------------------
    #pragma unroll
    for (int off = 16; off >= 1; off >>= 1) nll += __shfl_xor(nll, off);
    if (lane == 0) sPart[wave] = nll;
    __syncthreads();
    if (tid == 0) {
        float s = 0.f;
        #pragma unroll
        for (int w = 0; w < 8; ++w) s += sPart[w];
        partial[blockIdx.x] = s;
    }
}

// ---------------------------------------------------------------------------
// Final reduction: fixed-order sum of block partials, scale by 1/N.
// ---------------------------------------------------------------------------
__global__ __launch_bounds__(PL_THREADS)
void ploss_reduce_83133386981798(const float* __restrict__ partial, int nb,
                                 float inv_n, float* __restrict__ out)
{
    __shared__ float s[PL_THREADS];
    float a = 0.f;
    for (int i = threadIdx.x; i < nb; i += PL_THREADS) a += partial[i];
    s[threadIdx.x] = a;
    __syncthreads();
    for (int off = PL_THREADS / 2; off >= 1; off >>= 1) {
        if ((int)threadIdx.x < off) s[threadIdx.x] += s[threadIdx.x + off];
        __syncthreads();
    }
    if (threadIdx.x == 0) out[0] = s[0] * inv_n;
}

extern "C" void kernel_launch(void* const* d_in, const int* in_sizes, int n_in,
                              void* d_out, int out_size, void* d_ws, size_t ws_size,
                              hipStream_t stream)
{
    const float* outputs = (const float*)d_in[0];   // [N, 128] f32
    const int*   labels  = (const int*)d_in[1];     // [N] i32
    const float* protos  = (const float*)d_in[2];   // [128, 128] f32
    const int*   kptr    = (const int*)d_in[3];     // scalar k

    const int nrows = in_sizes[0] / PL_C;           // N = 400000
    const int nb    = (nrows + PL_ROWS_PER_BLOCK - 1) / PL_ROWS_PER_BLOCK;  // 3125

    float* partial = (float*)d_ws;                  // nb floats of scratch

    ploss_main_83133386981798<<<nb, PL_THREADS, 0, stream>>>(
        outputs, labels, protos, kptr, partial, nrows);

    ploss_reduce_83133386981798<<<1, PL_THREADS, 0, stream>>>(
        partial, nb, 1.0f / (float)nrows, (float*)d_out);
}
